// Model_52553219833973
// MI455X (gfx1250) — compile-verified, matrix operand order checked by reading, not run on hardware
//
#include <hip/hip_runtime.h>
#include <cstdint>
#include <cstddef>

// ---------------------------------------------------------------------------
// CDNA5 (gfx1250): wave32, WMMA bf16 16x16x32 f32-acc, TDM tile staging.
// ---------------------------------------------------------------------------

typedef __attribute__((ext_vector_type(16))) __bf16 v16bf;
typedef __attribute__((ext_vector_type(8)))  __bf16 v8bf;
typedef __attribute__((ext_vector_type(8)))  float  v8f;

#define LALPHA 0.2f

#if defined(__AMDGCN__) && __has_builtin(__builtin_amdgcn_tensor_load_to_lds) && \
    __has_builtin(__builtin_amdgcn_s_wait_tensorcnt)
#define USE_TDM 1
#else
#define USE_TDM 0
#endif

__device__ __forceinline__ v8f zero8() {
  v8f z;
#pragma unroll
  for (int i = 0; i < 8; ++i) z[i] = 0.f;
  return z;
}

__device__ __forceinline__ float sigmoidf_(float x) { return 1.f / (1.f + __expf(-x)); }

// 16x32 bf16 WMMA fragment from row-major bf16 (A: [M][K]; B staged as [N][K]).
// ISA 7.12.2: lane L holds row r=L&15; kbase=(L<16)?0:8;
// element e -> K = kbase + ((e>=8)?16:0) + (e&7). Two 16B chunks per lane.
__device__ __forceinline__ v16bf frag_ld(const __bf16* base, int stride, int lane) {
  const int r  = lane & 15;
  const int kb = (lane < 16) ? 0 : 8;
  const __bf16* p = base + r * stride + kb;
  v8bf lo = *(const v8bf*)(p);
  v8bf hi = *(const v8bf*)(p + 16);
  return __builtin_shufflevector(lo, hi, 0,1,2,3,4,5,6,7,8,9,10,11,12,13,14,15);
}

// Vectorized A staging: requires M % 64 == 0 and K % 8 == 0 (true for all uses).
__device__ __forceinline__ void stage_a_tile(const float* __restrict__ A,
                                             __bf16 (*As)[32],
                                             int bm, int K, int k0, int tid) {
  int r = tid >> 2, ks = (tid & 3) * 8;
  const float* ap = A + (size_t)(bm + r) * K + k0 + ks;
  v8bf av;
  if (k0 + ks < K) {
    float4 f0 = *(const float4*)ap;
    float4 f1 = *(const float4*)(ap + 4);
    av[0] = (__bf16)f0.x; av[1] = (__bf16)f0.y; av[2] = (__bf16)f0.z; av[3] = (__bf16)f0.w;
    av[4] = (__bf16)f1.x; av[5] = (__bf16)f1.y; av[6] = (__bf16)f1.z; av[7] = (__bf16)f1.w;
  } else {
#pragma unroll
    for (int i = 0; i < 8; ++i) av[i] = (__bf16)0.f;
  }
  *(v8bf*)&As[r][ks] = av;
}

// Vectorized B staging from pre-converted bf16 [N][K]: requires K % 16 == 0.
__device__ __forceinline__ void stage_b_tile(const __bf16* __restrict__ Bnk,
                                             __bf16 (*Bs)[32],
                                             int bn, int N, int K, int k0, int tid) {
  int n = tid >> 1, half = (tid & 1) * 16;
  const __bf16* src = Bnk + (size_t)(bn + n) * K + k0 + half;
  v8bf z;
#pragma unroll
  for (int i = 0; i < 8; ++i) z[i] = (__bf16)0.f;
  bool ok = (bn + n < N) && (k0 + half < K);
  if (ok && k0 + 32 < K) __builtin_prefetch(src + 32, 0, 1);
  v8bf lo = ok ? *(const v8bf*)src : z;
  v8bf hi = ok ? *(const v8bf*)(src + 8) : z;
  *(v8bf*)&Bs[n][half] = lo;
  *(v8bf*)&Bs[n][half + 8] = hi;
}

#if USE_TDM
typedef __attribute__((ext_vector_type(4))) unsigned int tdm_ui4;
typedef __attribute__((ext_vector_type(8))) int tdm_i8;
typedef __attribute__((ext_vector_type(4))) int tdm_i4;

// Issue one TDM 2D tile load: 128 rows x 32 bf16 (64B/row), row stride K elems,
// into contiguous LDS (rows packed at 64B). Issued by one wave; EXEC ignored.
__device__ __forceinline__ void tdm_load_b_tile(const __bf16* gsrc, void* ldst, int Kelems) {
  unsigned long long ga = (unsigned long long)(size_t)gsrc;
  unsigned lds = (unsigned)(size_t)ldst;  // low 32 bits of flat LDS addr = LDS offset
  tdm_ui4 g0 = {
      1u,                                                  // count=1, user desc
      lds,                                                 // lds_addr
      (unsigned)(ga & 0xFFFFFFFFu),                        // global_addr[31:0]
      (unsigned)((ga >> 32) & 0x1FFFFFFu) | (2u << 30)     // addr[56:32] | type=2
  };
  tdm_i8 g1 = {
      (int)(1u << 16),                                     // data_size=1 (2 bytes)
      (int)((32u & 0xFFFFu) << 16),                        // tensor_dim0=32 (lo16)
      (int)((128u & 0xFFFFu) << 16),                       // dim0 hi16=0 | tensor_dim1=128 lo16
      (int)(32u << 16),                                    // dim1 hi16=0 | tile_dim0=32
      (int)(128u),                                         // tile_dim1=128, tile_dim2=0
      (int)(unsigned)Kelems,                               // tensor_dim0_stride lo32
      0, 0
  };
  tdm_i4 z4 = {0, 0, 0, 0};
  tdm_i8 z8 = {0, 0, 0, 0, 0, 0, 0, 0};
  // 6-arg toolchain form: (g0, g1, g2, g3, g4, cpol)
  __builtin_amdgcn_tensor_load_to_lds(g0, g1, z4, z4, z8, 0);
}
#endif

// ---------------------------------------------------------------------------
// WMMA GEMM: C[M,N] = act(A[M,K] @ B + bias) (+C if accum). B pre-converted
// bf16 [N][K]. Block 256 thr = 8 waves (2x4), tile 64x128x32.
// Requires M%64==0, K%16==0. ACT: 0=none 1=sigmoid 2=leaky(0.2)
// ---------------------------------------------------------------------------
template <int ACT>
__global__ void __launch_bounds__(256) gemm_wmma(
    const float* __restrict__ A, const __bf16* __restrict__ Bnk,
    const float* __restrict__ bias, float* __restrict__ C,
    int M, int N, int K, int accum) {
  __shared__ __bf16 As[64][32];
  __shared__ __bf16 Bs[128][32];   // [n][k]
  const int tid  = threadIdx.x;
  const int lane = tid & 31;
  const int wave = tid >> 5;
  const int wm = wave >> 2, wn = wave & 3;
  const int bm = blockIdx.y * 64, bn = blockIdx.x * 128;
  v8f acc[2][2];
  acc[0][0] = zero8(); acc[0][1] = zero8(); acc[1][0] = zero8(); acc[1][1] = zero8();

  for (int k0 = 0; k0 < K; k0 += 32) {
    stage_a_tile(A, As, bm, K, k0, tid);
    bool bfull = (bn + 128 <= N) && (k0 + 32 <= K);
#if USE_TDM
    if (bfull) {
      if (wave == 0) {
        tdm_load_b_tile(Bnk + (size_t)bn * K + k0, &Bs[0][0], K);
        __builtin_amdgcn_s_wait_tensorcnt(0);
      }
    } else {
      stage_b_tile(Bnk, Bs, bn, N, K, k0, tid);
    }
#else
    (void)bfull;
    stage_b_tile(Bnk, Bs, bn, N, K, k0, tid);
#endif
    __syncthreads();
    v16bf af[2], bfr[2];
#pragma unroll
    for (int mi = 0; mi < 2; ++mi) af[mi]  = frag_ld(&As[wm * 32 + mi * 16][0], 32, lane);
#pragma unroll
    for (int ni = 0; ni < 2; ++ni) bfr[ni] = frag_ld(&Bs[wn * 32 + ni * 16][0], 32, lane);
#pragma unroll
    for (int mi = 0; mi < 2; ++mi)
#pragma unroll
      for (int ni = 0; ni < 2; ++ni)
        acc[mi][ni] = __builtin_amdgcn_wmma_f32_16x16x32_bf16(
            false, af[mi], false, bfr[ni], (short)0, acc[mi][ni], false, false);
    __syncthreads();
  }
  const int rb = (lane < 16) ? 0 : 8, cl = lane & 15;
#pragma unroll
  for (int mi = 0; mi < 2; ++mi)
#pragma unroll
    for (int ni = 0; ni < 2; ++ni)
#pragma unroll
      for (int r = 0; r < 8; ++r) {
        int row = bm + wm * 32 + mi * 16 + rb + r;
        int col = bn + wn * 32 + ni * 16 + cl;
        if (row < M && col < N) {
          float v = acc[mi][ni][r];
          if (bias) v += bias[col];
          if (ACT == 1) v = sigmoidf_(v);
          else if (ACT == 2) v = (v > 0.f) ? v : LALPHA * v;
          size_t idx = (size_t)row * N + col;
          if (accum) v += C[idx];
          C[idx] = v;
        }
      }
}

// ---------------------------------------------------------------------------
// Fused GLU GEMM: C = (A@W0 + b0) * sigmoid(A@W1 + b1). W0/W1 bf16 [N][K].
// ---------------------------------------------------------------------------
__global__ void __launch_bounds__(256) glu_gemm_wmma(
    const float* __restrict__ A, const __bf16* __restrict__ W0,
    const __bf16* __restrict__ W1, const float* __restrict__ b0,
    const float* __restrict__ b1, float* __restrict__ C,
    int M, int N, int K) {
  __shared__ __bf16 As[64][32];
  __shared__ __bf16 Bs0[128][32];
  __shared__ __bf16 Bs1[128][32];
  const int tid  = threadIdx.x;
  const int lane = tid & 31;
  const int wave = tid >> 5;
  const int wm = wave >> 2, wn = wave & 3;
  const int bm = blockIdx.y * 64, bn = blockIdx.x * 128;
  v8f a0[2][2], a1[2][2];
#pragma unroll
  for (int mi = 0; mi < 2; ++mi)
#pragma unroll
    for (int ni = 0; ni < 2; ++ni) { a0[mi][ni] = zero8(); a1[mi][ni] = zero8(); }

  for (int k0 = 0; k0 < K; k0 += 32) {
    stage_a_tile(A, As, bm, K, k0, tid);
    bool bfull = (bn + 128 <= N) && (k0 + 32 <= K);
#if USE_TDM
    if (bfull) {
      if (wave == 0) {
        tdm_load_b_tile(W0 + (size_t)bn * K + k0, &Bs0[0][0], K);
        tdm_load_b_tile(W1 + (size_t)bn * K + k0, &Bs1[0][0], K);
        __builtin_amdgcn_s_wait_tensorcnt(0);
      }
    } else {
      stage_b_tile(W0, Bs0, bn, N, K, k0, tid);
      stage_b_tile(W1, Bs1, bn, N, K, k0, tid);
    }
#else
    (void)bfull;
    stage_b_tile(W0, Bs0, bn, N, K, k0, tid);
    stage_b_tile(W1, Bs1, bn, N, K, k0, tid);
#endif
    __syncthreads();
    v16bf af[2];
#pragma unroll
    for (int mi = 0; mi < 2; ++mi) af[mi] = frag_ld(&As[wm * 32 + mi * 16][0], 32, lane);
#pragma unroll
    for (int ni = 0; ni < 2; ++ni) {
      v16bf bf0 = frag_ld(&Bs0[wn * 32 + ni * 16][0], 32, lane);
      v16bf bf1 = frag_ld(&Bs1[wn * 32 + ni * 16][0], 32, lane);
#pragma unroll
      for (int mi = 0; mi < 2; ++mi) {
        a0[mi][ni] = __builtin_amdgcn_wmma_f32_16x16x32_bf16(
            false, af[mi], false, bf0, (short)0, a0[mi][ni], false, false);
        a1[mi][ni] = __builtin_amdgcn_wmma_f32_16x16x32_bf16(
            false, af[mi], false, bf1, (short)0, a1[mi][ni], false, false);
      }
    }
    __syncthreads();
  }
  const int rb = (lane < 16) ? 0 : 8, cl = lane & 15;
#pragma unroll
  for (int mi = 0; mi < 2; ++mi)
#pragma unroll
    for (int ni = 0; ni < 2; ++ni)
#pragma unroll
      for (int r = 0; r < 8; ++r) {
        int row = bm + wm * 32 + mi * 16 + rb + r;
        int col = bn + wn * 32 + ni * 16 + cl;
        if (row < M && col < N) {
          float u = a0[mi][ni][r] + b0[col];
          float g = a1[mi][ni][r] + b1[col];
          C[(size_t)row * N + col] = u * sigmoidf_(g);
        }
      }
}

// ---------------------------------------------------------------------------
// GRU: single resident workgroup, 1024 thr = 32 waves. h + gates in LDS
// (288 KB of the 320 KB WGP LDS); whh streamed bf16 (L2 resident).
// ---------------------------------------------------------------------------
__global__ void __launch_bounds__(1024) gru_kernel(
    const float* __restrict__ gi_all,   // (512, 32, 1536), bih folded in
    const __bf16* __restrict__ whhbf,   // (1536, 512) row-major bf16 = [N][K]
    const float* __restrict__ bhh,
    const float* __restrict__ wk, const float* __restrict__ wq,
    float* __restrict__ key, float* __restrict__ query) {  // (32,512) each
  extern __shared__ char smem[];
  float*  hS  = (float*)smem;                                 // 32*512 f32
  __bf16* hbf = (__bf16*)(smem + 32 * 512 * 4);               // 32*512 bf16
  float*  ghs = (float*)(smem + 32 * 512 * 4 + 32 * 512 * 2); // 32*1536 f32
  const int tid = threadIdx.x, lane = tid & 31, wave = tid >> 5;
  for (int i = tid; i < 32 * 512; i += 1024) { hS[i] = 0.f; hbf[i] = (__bf16)0.f; }
  __syncthreads();
  const int jbase = wave * 48;

  for (int n = 0; n < 512; ++n) {
    v8f acc[2][3];
#pragma unroll
    for (int mi = 0; mi < 2; ++mi)
#pragma unroll
      for (int nj = 0; nj < 3; ++nj) acc[mi][nj] = zero8();
    for (int k0 = 0; k0 < 512; k0 += 32) {
      v16bf af[2];
#pragma unroll
      for (int mi = 0; mi < 2; ++mi) af[mi] = frag_ld(hbf + (size_t)(mi * 16) * 512 + k0, 512, lane);
#pragma unroll
      for (int nj = 0; nj < 3; ++nj) {
        v16bf bf = frag_ld(whhbf + (size_t)(jbase + nj * 16) * 512 + k0, 512, lane);
#pragma unroll
        for (int mi = 0; mi < 2; ++mi)
          acc[mi][nj] = __builtin_amdgcn_wmma_f32_16x16x32_bf16(
              false, af[mi], false, bf, (short)0, acc[mi][nj], false, false);
      }
    }
    const int rb = (lane < 16) ? 0 : 8, cl = lane & 15;
#pragma unroll
    for (int mi = 0; mi < 2; ++mi)
#pragma unroll
      for (int nj = 0; nj < 3; ++nj)
#pragma unroll
        for (int r = 0; r < 8; ++r) {
          int b = mi * 16 + rb + r;
          int j = jbase + nj * 16 + cl;
          ghs[b * 1536 + j] = acc[mi][nj][r] + bhh[j];
        }
    __syncthreads();
    const float* gip = gi_all + (size_t)n * 32 * 1536;
#pragma unroll 4
    for (int it = 0; it < 16; ++it) {
      int idx = tid + it * 1024;
      int b = idx >> 9, j = idx & 511;
      float ir  = gip[b * 1536 + j];
      float iz  = gip[b * 1536 + 512 + j];
      float inn = gip[b * 1536 + 1024 + j];
      float hr = ghs[b * 1536 + j];
      float hz = ghs[b * 1536 + 512 + j];
      float hn = ghs[b * 1536 + 1024 + j];
      float r = sigmoidf_(ir + hr);
      float z = sigmoidf_(iz + hz);
      float nn = tanhf(inn + r * hn);
      float h = (1.f - z) * nn + z * hS[idx];
      hS[idx] = h;
      hbf[idx] = (__bf16)h;
    }
    __syncthreads();
    {
      int b = wave;
      float sk = 0.f, sq = 0.f;
      for (int c = lane; c < 512; c += 32) {
        float h = hS[b * 512 + c];
        sk += h * wk[c];
        sq += h * wq[c];
      }
#pragma unroll
      for (int off = 16; off > 0; off >>= 1) {
        sk += __shfl_down(sk, off, 32);
        sq += __shfl_down(sq, off, 32);
      }
      if (lane == 0) { key[b * 512 + n] = sk; query[b * 512 + n] = sq; }
    }
    __syncthreads();
  }
}

// ---------------------------------------------------------------------------
// Prep / elementwise / attention kernels
// ---------------------------------------------------------------------------
__global__ void k_xpose(const float* __restrict__ x, float* __restrict__ xT,
                        float* __restrict__ xbnt) {
  int idx = blockIdx.x * 256 + threadIdx.x;
  if (idx >= 32 * 48 * 512) return;
  int n = idx % 512, t = (idx / 512) % 48, b = idx / (512 * 48);
  float v = x[idx];
  xT[n * 1536 + b * 48 + t] = v;            // (n, b, t): GRU-input GEMM A
  xbnt[(b * 512 + n) * 48 + t] = v;         // (b, n, t): "short" GEMM A
}

__global__ void k_cvt_bf16(const float* __restrict__ s, __bf16* __restrict__ d, int n) {
  int idx = blockIdx.x * 256 + threadIdx.x;
  if (idx < n) d[idx] = (__bf16)s[idx];
}

// f32 weight [K][N] row-major -> bf16 [N][K]
__global__ void k_w2bf(const float* __restrict__ w, __bf16* __restrict__ o, int K, int N) {
  int idx = blockIdx.x * 256 + threadIdx.x;
  if (idx >= K * N) return;
  int k = idx / N, n = idx % N;
  o[(size_t)n * K + k] = (__bf16)w[idx];
}

__global__ void att_softmax(const float* __restrict__ key, const float* __restrict__ query,
                            float* __restrict__ att_raw) {
  int i = blockIdx.x, t = threadIdx.x;  // 512 blocks x 256 thr
  __shared__ float row[512];
  __shared__ float accv[512];
  __shared__ float red[256];
  accv[t] = 0.f; accv[t + 256] = 0.f;
  __syncthreads();
  for (int b = 0; b < 32; ++b) {
    float ki = key[b * 512 + i];
#pragma unroll
    for (int jj = 0; jj < 2; ++jj) {
      int j = t + jj * 256;
      float d = ki + query[b * 512 + j];
      row[j] = (d > 0.f) ? d : LALPHA * d;
    }
    __syncthreads();
    red[t] = fmaxf(row[t], row[t + 256]);
    __syncthreads();
    for (int s = 128; s > 0; s >>= 1) { if (t < s) red[t] = fmaxf(red[t], red[t + s]); __syncthreads(); }
    float mx = red[0];
    __syncthreads();
    float e0 = __expf(row[t] - mx), e1 = __expf(row[t + 256] - mx);
    red[t] = e0 + e1;
    __syncthreads();
    for (int s = 128; s > 0; s >>= 1) { if (t < s) red[t] += red[t + s]; __syncthreads(); }
    float inv = 1.f / red[0];
    accv[t] += e0 * inv; accv[t + 256] += e1 * inv;
    __syncthreads();
  }
  att_raw[i * 512 + t] = accv[t] * (1.f / 32.f);
  att_raw[i * 512 + t + 256] = accv[t + 256] * (1.f / 32.f);
}

__global__ void att_sym(const float* __restrict__ raw, float* __restrict__ att_out,
                        float* __restrict__ deg, float* __restrict__ dis) {
  int i = blockIdx.x, t = threadIdx.x;
  __shared__ float red[256];
  float s = 0.f;
#pragma unroll
  for (int jj = 0; jj < 2; ++jj) {
    int j = t + jj * 256;
    float v = 0.5f * (raw[i * 512 + j] + raw[j * 512 + i]);
    att_out[i * 512 + j] = v;
    s += v;
  }
  red[t] = s;
  __syncthreads();
  for (int st = 128; st > 0; st >>= 1) { if (t < st) red[t] += red[t + st]; __syncthreads(); }
  if (t == 0) { float d = fmaxf(red[0], 1e-7f); deg[i] = d; dis[i] = rsqrtf(d); }
}

__global__ void lap_build(const float* __restrict__ att, const float* __restrict__ deg,
                          const float* __restrict__ dis, float* __restrict__ mull) {
  int idx = blockIdx.x * 256 + threadIdx.x;
  if (idx >= 512 * 512) return;
  int i = idx / 512, j = idx % 512;
  float l = dis[i] * (((i == j) ? deg[i] : 0.f) - att[idx]) * dis[j];
  mull[idx] = (i == j) ? 1.f : 0.f;   // t0 = I
  mull[262144 + idx] = l;             // t1 = lap
}

// C = 2*A@B - D  (512^3, plain f32 — tiny, full precision)
__global__ void cheb_gemm(const float* __restrict__ A, const float* __restrict__ Bm,
                          const float* __restrict__ D, float* __restrict__ C) {
  __shared__ float As[16][16], Bs[16][16];
  int tx = threadIdx.x, ty = threadIdx.y;
  int row = blockIdx.y * 16 + ty, col = blockIdx.x * 16 + tx;
  float acc = 0.f;
  for (int k0 = 0; k0 < 512; k0 += 16) {
    As[ty][tx] = A[row * 512 + k0 + tx];
    Bs[ty][tx] = Bm[(k0 + ty) * 512 + col];
    __syncthreads();
#pragma unroll
    for (int k = 0; k < 16; ++k) acc += As[ty][k] * Bs[k][tx];
    __syncthreads();
  }
  C[row * 512 + col] = 2.f * acc - D[row * 512 + col];
}

// length-4 FFT over k, flatten to (b*512+n, k*48+t). Imag of F0,F2 is zero.
__global__ void fftflat(const float* __restrict__ g, float* __restrict__ fr,
                        float* __restrict__ fi) {
  int idx = blockIdx.x * 256 + threadIdx.x;
  if (idx >= 32 * 512 * 48) return;
  int t = idx % 48, n = (idx / 48) % 512, b = idx / (48 * 512);
  const size_t PLN = (size_t)512 * 1536;
  size_t base = (size_t)n * 1536 + (size_t)b * 48 + t;
  float x0 = g[base], x1 = g[PLN + base], x2 = g[2 * PLN + base], x3 = g[3 * PLN + base];
  float F0 = x0 + x1 + x2 + x3;
  float F2 = x0 - x1 + x2 - x3;
  float R1 = x0 - x2;
  float I1 = -(x1 - x3);
  size_t rb = ((size_t)b * 512 + n) * 192;
  fr[rb + t] = F0;  fr[rb + 48 + t] = R1;  fr[rb + 96 + t] = F2;  fr[rb + 144 + t] = R1;
  fi[rb + t] = 0.f; fi[rb + 48 + t] = I1;  fi[rb + 96 + t] = 0.f; fi[rb + 144 + t] = -I1;
}

// irfft(n=4) over k: crops to X0..X2.
__global__ void irfft4(const float* __restrict__ re, const float* __restrict__ im,
                       float* __restrict__ sp) {
  int idx = blockIdx.x * 256 + threadIdx.x;
  if (idx >= 16384 * 240) return;
  int p = idx % 240; size_t row = idx / 240;
  float R0 = re[row * 960 + p], R1 = re[row * 960 + 240 + p], R2 = re[row * 960 + 480 + p];
  float Im1 = im[row * 960 + 240 + p];
  sp[row * 960 + p]       = 0.25f * (R0 + R2 + 2.f * R1);
  sp[row * 960 + 240 + p] = 0.25f * (R0 - R2 - 2.f * Im1);
  sp[row * 960 + 480 + p] = 0.25f * (R0 + R2 - 2.f * R1);
  sp[row * 960 + 720 + p] = 0.25f * (R0 - R2 + 2.f * Im1);
}

// block-2 input sigmoid(bc - short), written straight to bf16 [col=(b,t)][n]
__global__ void sigdiff(const float* __restrict__ bc, const float* __restrict__ sh,
                        __bf16* __restrict__ x2bf) {
  int idx = blockIdx.x * 256 + threadIdx.x;
  if (idx >= 16384 * 48) return;
  int t = idx % 48; int row = idx / 48;
  int b = row >> 9, n = row & 511;
  x2bf[(size_t)(b * 48 + t) * 512 + n] = (__bf16)sigmoidf_(bc[idx] - sh[idx]);
}

// (b*512+n, 12) -> forecast (b, 12, 512)
__global__ void out_tr(const float* __restrict__ ob, float* __restrict__ out) {
  int idx = blockIdx.x * 256 + threadIdx.x;
  if (idx >= 16384 * 12) return;
  int hz = idx % 12; int row = idx / 12;
  int b = row >> 9, n = row & 511;
  out[b * 6144 + hz * 512 + n] = ob[idx];
}

// ---------------------------------------------------------------------------
// Host orchestration
// ---------------------------------------------------------------------------
static inline int cdiv_(int a, int b) { return (a + b - 1) / b; }

extern "C" void kernel_launch(void* const* d_in, const int* in_sizes, int n_in,
                              void* d_out, int out_size, void* d_ws, size_t ws_size,
                              hipStream_t stream) {
  (void)in_sizes; (void)n_in; (void)out_size; (void)ws_size;
  const float* x     = (const float*)d_in[0];
  const float* wk    = (const float*)d_in[1];
  const float* wq    = (const float*)d_in[2];
  const float* wih   = (const float*)d_in[3];
  const float* whh   = (const float*)d_in[4];
  const float* bih   = (const float*)d_in[5];
  const float* bhh   = (const float*)d_in[6];
  const float* blkw  = (const float*)d_in[7];
  const float* fcw   = (const float*)d_in[8];
  const float* fcb   = (const float*)d_in[9];
  const float* frw   = (const float*)d_in[10];
  const float* frb   = (const float*)d_in[11];
  const float* g0w   = (const float*)d_in[12];
  const float* g0b   = (const float*)d_in[13];
  const float* g12w  = (const float*)d_in[14];
  const float* g12b  = (const float*)d_in[15];
  const float* backw = (const float*)d_in[16];
  const float* backb = (const float*)d_in[17];
  const float* bscw  = (const float*)d_in[18];
  const float* bscb  = (const float*)d_in[19];
  const float* fc1w  = (const float*)d_in[20];
  const float* fc1b  = (const float*)d_in[21];
  const float* fc2w  = (const float*)d_in[22];
  const float* fc2b  = (const float*)d_in[23];
  float* out = (float*)d_out;

  char* ws = (char*)d_ws;
  size_t cur = 0;
  auto alloc = [&](size_t bytes) -> char* {
    char* p = ws + cur;
    cur = (cur + bytes + 255) & ~(size_t)255;
    return p;
  };
  float*  xT     = (float*)alloc((size_t)786432 * 4);
  float*  xbnt   = (float*)alloc((size_t)786432 * 4);
  __bf16* xbf    = (__bf16*)alloc((size_t)786432 * 2);   // x as [(b,t)][n] bf16
  __bf16* x2bf   = (__bf16*)alloc((size_t)786432 * 2);
  __bf16* wihbf  = (__bf16*)alloc((size_t)73728 * 2);    // [1536][48] = [N][K]
  __bf16* whhbf  = (__bf16*)alloc((size_t)786432 * 2);
  __bf16* g0bf   = (__bf16*)alloc((size_t)8 * 184320 * 2);
  __bf16* g12bf  = (__bf16*)alloc((size_t)16 * 921600 * 2);
  __bf16* blkbf  = (__bf16*)alloc((size_t)2 * 230400 * 2);
  __bf16* fcwbf  = (__bf16*)alloc((size_t)2 * 57600 * 2);
  __bf16* frwbf  = (__bf16*)alloc((size_t)2 * 11520 * 2);
  __bf16* backbf = (__bf16*)alloc((size_t)11520 * 2);
  __bf16* bscbf  = (__bf16*)alloc((size_t)2304 * 2);
  __bf16* fc1bf  = (__bf16*)alloc((size_t)2304 * 2);
  __bf16* fc2bf  = (__bf16*)alloc((size_t)576 * 2);
  float*  giAll  = (float*)alloc((size_t)25165824 * 4);
  float*  key    = (float*)alloc((size_t)16384 * 4);
  float*  query  = (float*)alloc((size_t)16384 * 4);
  float*  araw   = (float*)alloc((size_t)262144 * 4);
  float*  deg    = (float*)alloc((size_t)512 * 4);
  float*  dis    = (float*)alloc((size_t)512 * 4);
  float*  mull   = (float*)alloc((size_t)1048576 * 4);
  float*  gfted  = (float*)alloc((size_t)3145728 * 4);
  float*  fre    = (float*)alloc((size_t)3145728 * 4);
  float*  fim    = (float*)alloc((size_t)3145728 * 4);
  float*  bigA   = (float*)alloc((size_t)15728640 * 4);
  float*  bigB   = (float*)alloc((size_t)15728640 * 4);
  float*  bigC   = (float*)alloc((size_t)15728640 * 4);
  float*  igf    = (float*)alloc((size_t)3932160 * 4);
  float*  fsb    = (float*)alloc((size_t)3932160 * 4);
  float*  f01    = (float*)alloc((size_t)786432 * 4);
  float*  bcb    = (float*)alloc((size_t)786432 * 4);
  float*  shb    = (float*)alloc((size_t)786432 * 4);
  float*  hb     = (float*)alloc((size_t)786432 * 4);
  float*  ob     = (float*)alloc((size_t)196608 * 4);
  float*  attO   = out + 196608;

  auto W2BF = [&](const float* w, __bf16* o, int K, int N) {
    k_w2bf<<<cdiv_(K * N, 256), 256, 0, stream>>>(w, o, K, N);
  };

  // --- prep: transposes + bf16 weight pre-conversion ------------------------
  k_xpose<<<cdiv_(786432, 256), 256, 0, stream>>>(x, xT, xbnt);
  k_cvt_bf16<<<cdiv_(73728, 256), 256, 0, stream>>>(wih, wihbf, 73728);      // already [N][K]
  k_cvt_bf16<<<cdiv_(786432, 256), 256, 0, stream>>>(whh, whhbf, 786432);    // [1536][512]
  k_cvt_bf16<<<cdiv_(786432, 256), 256, 0, stream>>>(x, xbf, 786432);        // x rows (b,t) x 512
  for (int s = 0; s < 8; ++s)
    W2BF(g0w + (size_t)s * 184320, g0bf + (size_t)s * 184320, 192, 960);
  for (int s = 0; s < 16; ++s)
    W2BF(g12w + (size_t)s * 921600, g12bf + (size_t)s * 921600, 960, 960);
  for (int i = 0; i < 2; ++i) {
    W2BF(blkw + (size_t)i * 230400, blkbf + (size_t)i * 230400, 960, 240);
    W2BF(fcw + (size_t)i * 57600, fcwbf + (size_t)i * 57600, 240, 240);
    W2BF(frw + (size_t)i * 11520, frwbf + (size_t)i * 11520, 240, 48);
  }
  W2BF(backw, backbf, 240, 48);
  W2BF(bscw, bscbf, 48, 48);
  W2BF(fc1w, fc1bf, 48, 48);
  W2BF(fc2w, fc2bf, 48, 12);

  // gi_all = x_seq @ wih^T + bih : (16384,48)@(48,1536)
  gemm_wmma<0><<<dim3(12, 256), 256, 0, stream>>>(xT, wihbf, bih, giAll, 16384, 1536, 48, 0);

  // --- GRU (serial chain, one resident WGP) ---------------------------------
  const size_t GRU_SMEM = 32 * 512 * 4 + 32 * 512 * 2 + 32 * 1536 * 4;  // 288 KB
  gru_kernel<<<1, 1024, GRU_SMEM, stream>>>(giAll, whhbf, bhh, wk, wq, key, query);

  // --- attention + Laplacian + Chebyshev ------------------------------------
  att_softmax<<<512, 256, 0, stream>>>(key, query, araw);
  att_sym<<<512, 256, 0, stream>>>(araw, attO, deg, dis);
  lap_build<<<1024, 256, 0, stream>>>(attO, deg, dis, mull);
  const size_t L = 262144;
  cheb_gemm<<<dim3(32, 32), dim3(16, 16), 0, stream>>>(mull + L, mull + L, mull, mull + 2 * L);
  cheb_gemm<<<dim3(32, 32), dim3(16, 16), 0, stream>>>(mull + L, mull + 2 * L, mull + L, mull + 3 * L);

  // --- two blocks -----------------------------------------------------------
  auto G0Wp  = [&](int i, int ri, int w) { return g0bf  + ((size_t)((i * 2 + ri) * 2 + w)) * 184320; };
  auto G0B   = [&](int i, int ri, int w) { return g0b   + ((size_t)((i * 2 + ri) * 2 + w)) * 960; };
  auto G12Wp = [&](int i, int l, int ri, int w) { return g12bf + ((size_t)(((i * 2 + l) * 2 + ri) * 2 + w)) * 921600; };
  auto G12B  = [&](int i, int l, int ri, int w) { return g12b  + ((size_t)(((i * 2 + l) * 2 + ri) * 2 + w)) * 960; };

  for (int i = 0; i < 2; ++i) {
    const __bf16* xin_nk = (i == 0) ? xbf : x2bf;
    for (int k = 0; k < 4; ++k)
      gemm_wmma<0><<<dim3(12, 8), 256, 0, stream>>>(
          mull + (size_t)k * L, xin_nk, nullptr, gfted + (size_t)k * 786432, 512, 1536, 512, 0);
    fftflat<<<cdiv_(786432, 256), 256, 0, stream>>>(gfted, fre, fim);
    // real GLU chain: fre -> A -> B -> A
    glu_gemm_wmma<<<dim3(8, 256), 256, 0, stream>>>(fre,  G0Wp(i,0,0),    G0Wp(i,0,1),    G0B(i,0,0),    G0B(i,0,1),    bigA, 16384, 960, 192);
    glu_gemm_wmma<<<dim3(8, 256), 256, 0, stream>>>(bigA, G12Wp(i,0,0,0), G12Wp(i,0,0,1), G12B(i,0,0,0), G12B(i,0,0,1), bigB, 16384, 960, 960);
    glu_gemm_wmma<<<dim3(8, 256), 256, 0, stream>>>(bigB, G12Wp(i,1,0,0), G12Wp(i,1,0,1), G12B(i,1,0,0), G12B(i,1,0,1), bigA, 16384, 960, 960);
    // imag GLU chain: fim -> B -> C -> B
    glu_gemm_wmma<<<dim3(8, 256), 256, 0, stream>>>(fim,  G0Wp(i,1,0),    G0Wp(i,1,1),    G0B(i,1,0),    G0B(i,1,1),    bigB, 16384, 960, 192);
    glu_gemm_wmma<<<dim3(8, 256), 256, 0, stream>>>(bigB, G12Wp(i,0,1,0), G12Wp(i,0,1,1), G12B(i,0,1,0), G12B(i,0,1,1), bigC, 16384, 960, 960);
    glu_gemm_wmma<<<dim3(8, 256), 256, 0, stream>>>(bigC, G12Wp(i,1,1,0), G12Wp(i,1,1,1), G12B(i,1,1,0), G12B(i,1,1,1), bigB, 16384, 960, 960);
    // sp = irfft4(real=A, imag=B) -> C at (b,n, k*240+p)
    irfft4<<<cdiv_(16384 * 240, 256), 256, 0, stream>>>(bigA, bigB, bigC);
    // igfted = (16384,960) @ (960,240)
    gemm_wmma<0><<<dim3(2, 256), 256, 0, stream>>>(bigC, blkbf + (size_t)i * 230400, nullptr, igf, 16384, 240, 960, 0);
    // fs = sigmoid(igfted @ fcast_w + b)
    gemm_wmma<1><<<dim3(2, 256), 256, 0, stream>>>(igf, fcwbf + (size_t)i * 57600, fcb + (size_t)i * 240, fsb, 16384, 240, 240, 0);
    // forecast_i = fs @ fres_w + b   (block 1 accumulates)
    gemm_wmma<0><<<dim3(1, 256), 256, 0, stream>>>(fsb, frwbf + (size_t)i * 11520, frb + (size_t)i * 48, f01, 16384, 48, 240, i);
    if (i == 0) {
      gemm_wmma<0><<<dim3(1, 256), 256, 0, stream>>>(igf, backbf, backb, bcb, 16384, 48, 240, 0);
      gemm_wmma<0><<<dim3(1, 256), 256, 0, stream>>>(xbnt, bscbf, bscb, shb, 16384, 48, 48, 0);
      sigdiff<<<cdiv_(786432, 256), 256, 0, stream>>>(bcb, shb, x2bf);
    }
  }

  // --- head -----------------------------------------------------------------
  gemm_wmma<2><<<dim3(1, 256), 256, 0, stream>>>(f01, fc1bf, fc1b, hb, 16384, 48, 48, 0);
  gemm_wmma<0><<<dim3(1, 256), 256, 0, stream>>>(hb, fc2bf, fc2b, ob, 16384, 12, 48, 0);
  out_tr<<<cdiv_(196608, 256), 256, 0, stream>>>(ob, out);
}
